// PartSegHead_15942918602911
// MI455X (gfx1250) — matrix operand order repaired and checked by reading, not compiled, not run on hardware
//
#include <hip/hip_runtime.h>

typedef __attribute__((ext_vector_type(16))) __bf16 v16bf;
typedef __attribute__((ext_vector_type(8)))  __bf16 v8bf;
typedef __attribute__((ext_vector_type(8)))  float  v8f;

#define TILE_M   64
#define K1       512      // 451 padded to multiple of 32 (and of 128)
#define K2       256
#define LDA      520      // A-tile row stride (bf16): 512 + 8 -> 16B/row bank rotation
#define LDH      264      // H-tile row stride (bf16): 256 + 8
#define BN_EPS_F 1e-5f

#define SMEM_BYTES ((TILE_M*LDA + 2*TILE_M*LDH) * 2 + TILE_M*3*4*2)

__device__ __forceinline__ v16bf combine8(v8bf lo, v8bf hi) {
    return __builtin_shufflevector(lo, hi, 0,1,2,3,4,5,6,7,8,9,10,11,12,13,14,15);
}

// ---------------- prep: weights -> bf16 (padded), fold BN into alpha/beta ----
__global__ void prep_kernel(const float* __restrict__ w1, const float* __restrict__ w2,
                            const float* __restrict__ w3,
                            const float* __restrict__ b1, const float* __restrict__ g1,
                            const float* __restrict__ be1, const float* __restrict__ m1,
                            const float* __restrict__ v1,
                            const float* __restrict__ b2, const float* __restrict__ g2,
                            const float* __restrict__ be2, const float* __restrict__ m2,
                            const float* __restrict__ v2,
                            __bf16* __restrict__ w1bf, __bf16* __restrict__ w2bf,
                            __bf16* __restrict__ w3bf, float* __restrict__ ab) {
    int idx = blockIdx.x * 256 + threadIdx.x;
    const int R1 = 256 * K1, R2 = 256 * K2, R3 = 64 * K2;
    if (idx < R1) {
        int o = idx >> 9, k = idx & 511;                 // [256][512], pad k>=451
        w1bf[idx] = (k < 451) ? (__bf16)w1[o * 451 + k] : (__bf16)0.f;
    } else if ((idx -= R1) < R2) {
        w2bf[idx] = (__bf16)w2[idx];                     // [256][256]
    } else if ((idx -= R2) < R3) {
        int o = idx >> 8, k = idx & 255;                 // [64][256], pad o>=50
        w3bf[idx] = (o < 50) ? (__bf16)w3[o * 256 + k] : (__bf16)0.f;
    } else if ((idx -= R3) < 512) {
        if (idx < 256) {                                 // layer1 folded BN
            float s = g1[idx] * rsqrtf(v1[idx] + BN_EPS_F);
            ab[idx]       = s;
            ab[256 + idx] = b1[idx] * s + be1[idx] - m1[idx] * s;
        } else {                                         // layer2 folded BN
            int o = idx - 256;
            float s = g2[o] * rsqrtf(v2[o] + BN_EPS_F);
            ab[512 + o] = s;
            ab[768 + o] = b2[o] * s + be2[o] - m2[o] * s;
        }
    }
}

// ---------------- fused: knn-interp -> bf16 A tile -> 3x WMMA GEMM ----------
__global__ __launch_bounds__(256)
void partseg_fused(const float* __restrict__ feat,      // [16,64,384]
                   const float* __restrict__ centers,   // [16,64,3]
                   const float* __restrict__ pc,        // [16,8192,3]
                   const int*   __restrict__ category,  // [16]
                   const float* __restrict__ cat_embed, // [16,64]
                   const float* __restrict__ b3,        // [50]
                   const __bf16* __restrict__ w1bf,     // [256][512]
                   const __bf16* __restrict__ w2bf,     // [256][256]
                   const __bf16* __restrict__ w3bf,     // [64][256]
                   const float* __restrict__ ab,        // alpha1,beta1,alpha2,beta2
                   float* __restrict__ out)             // [16,8192,50]
{
    extern __shared__ char smem[];
    __bf16* As = (__bf16*)smem;                 // [64][520]
    __bf16* H1 = As + TILE_M * LDA;             // [64][264]
    __bf16* H2 = H1 + TILE_M * LDH;             // [64][264]
    float*  sW = (float*)(H2 + TILE_M * LDH);   // [64][3]
    int*    sI = (int*)(sW + TILE_M * 3);       // [64][3]

    const int tid = threadIdx.x;
    const int b   = blockIdx.x >> 7;            // 128 tiles of 64 points per batch
    const int n0  = (blockIdx.x & 127) * TILE_M;

    // ---- phase 1a: per-point top-3 of 64 centers (threads 0..63) ----
    if (tid < TILE_M) {
        const float* P = pc + ((size_t)b * 8192 + n0 + tid) * 3;
        float px = P[0], py = P[1], pz = P[2];
        float d0 = 3.4e38f, d1 = 3.4e38f, d2 = 3.4e38f;
        int   i0 = 0, i1 = 0, i2 = 0;
        const float* C = centers + (size_t)b * 64 * 3;
        for (int g = 0; g < 64; ++g) {
            float dx = px - C[g * 3 + 0];
            float dy = py - C[g * 3 + 1];
            float dz = pz - C[g * 3 + 2];
            float d = dx * dx + dy * dy + dz * dz;
            if (d < d0)      { d2 = d1; i2 = i1; d1 = d0; i1 = i0; d0 = d; i0 = g; }
            else if (d < d1) { d2 = d1; i2 = i1; d1 = d; i1 = g; }
            else if (d < d2) { d2 = d; i2 = g; }
        }
        float r0 = 1.f / (sqrtf(fmaxf(d0, 1e-12f)) + 1e-8f);
        float r1 = 1.f / (sqrtf(fmaxf(d1, 1e-12f)) + 1e-8f);
        float r2 = 1.f / (sqrtf(fmaxf(d2, 1e-12f)) + 1e-8f);
        float rs = 1.f / (r0 + r1 + r2);
        sW[tid * 3 + 0] = r0 * rs; sW[tid * 3 + 1] = r1 * rs; sW[tid * 3 + 2] = r2 * rs;
        sI[tid * 3 + 0] = i0;      sI[tid * 3 + 1] = i1;      sI[tid * 3 + 2] = i2;
    }
    __syncthreads();

    // ---- phase 1b: build bf16 A tile [64][512] (interp | cat | xyz | 0-pad) ----
    {
        const float* catRow = cat_embed + (size_t)category[b] * 64;
        for (int p = 0; p < TILE_M; ++p) {
            int   i0 = sI[p * 3 + 0], i1 = sI[p * 3 + 1], i2 = sI[p * 3 + 2];
            float w0 = sW[p * 3 + 0], wv1 = sW[p * 3 + 1], wv2 = sW[p * 3 + 2];
            const float* f0 = feat + ((size_t)b * 64 + i0) * 384;
            const float* f1 = feat + ((size_t)b * 64 + i1) * 384;
            const float* f2 = feat + ((size_t)b * 64 + i2) * 384;
            const float* P  = pc + ((size_t)b * 8192 + n0 + p) * 3;
            for (int d = tid; d < K1; d += 256) {
                float v;
                if (d < 384)      v = w0 * f0[d] + wv1 * f1[d] + wv2 * f2[d];
                else if (d < 448) v = catRow[d - 384];
                else if (d < 451) v = P[d - 448];
                else              v = 0.f;
                As[p * LDA + d] = (__bf16)v;
            }
        }
    }
    __syncthreads();

    const int wave = tid >> 5, lane = tid & 31;
    const int lh = lane >> 4, ll = lane & 15;   // half-wave, lane-in-half

    // ---- phase 2: GEMM1  H1[64][256] = relu(affine1(A[64][512] x W1^T)) ----
    for (int t = wave; t < 64; t += 8) {
        int mt = t & 3, nt = t >> 2;
        v8f acc = {};
        const __bf16* aRow = &As[(mt * 16 + ll) * LDA];
        const __bf16* bRow = w1bf + (size_t)(nt * 16 + ll) * K1 + lh * 8;
        #pragma unroll 4
        for (int k0 = 0; k0 < K1; k0 += 32) {
            v8bf alo = *(const v8bf*)(aRow + k0 + lh * 8);
            v8bf ahi = *(const v8bf*)(aRow + k0 + lh * 8 + 16);
            v8bf blo = *(const v8bf*)(bRow + k0);
            v8bf bhi = *(const v8bf*)(bRow + k0 + 16);
            acc = __builtin_amdgcn_wmma_f32_16x16x32_bf16(
                false, combine8(alo, ahi), false, combine8(blo, bhi),
                (short)0, acc, false, false);
        }
        int o = nt * 16 + ll;
        float alpha = ab[o], beta = ab[256 + o];
        #pragma unroll
        for (int j = 0; j < 8; ++j) {
            float y = fmaxf(acc[j] * alpha + beta, 0.f);
            H1[(mt * 16 + lh * 8 + j) * LDH + o] = (__bf16)y;
        }
    }
    __syncthreads();

    // ---- phase 3: GEMM2  H2[64][256] = relu(affine2(H1 x W2^T)) ----
    for (int t = wave; t < 64; t += 8) {
        int mt = t & 3, nt = t >> 2;
        v8f acc = {};
        const __bf16* aRow = &H1[(mt * 16 + ll) * LDH];
        const __bf16* bRow = w2bf + (size_t)(nt * 16 + ll) * K2 + lh * 8;
        #pragma unroll
        for (int k0 = 0; k0 < K2; k0 += 32) {
            v8bf alo = *(const v8bf*)(aRow + k0 + lh * 8);
            v8bf ahi = *(const v8bf*)(aRow + k0 + lh * 8 + 16);
            v8bf blo = *(const v8bf*)(bRow + k0);
            v8bf bhi = *(const v8bf*)(bRow + k0 + 16);
            acc = __builtin_amdgcn_wmma_f32_16x16x32_bf16(
                false, combine8(alo, ahi), false, combine8(blo, bhi),
                (short)0, acc, false, false);
        }
        int o = nt * 16 + ll;
        float alpha = ab[512 + o], beta = ab[768 + o];
        #pragma unroll
        for (int j = 0; j < 8; ++j) {
            float y = fmaxf(acc[j] * alpha + beta, 0.f);
            H2[(mt * 16 + lh * 8 + j) * LDH + o] = (__bf16)y;
        }
    }
    __syncthreads();

    // ---- phase 4: GEMM3  logits[64][50] = H2 x W3^T + b3 (write f32 out) ----
    for (int t = wave; t < 16; t += 8) {
        int mt = t & 3, nt = t >> 2;
        v8f acc = {};
        const __bf16* aRow = &H2[(mt * 16 + ll) * LDH];
        const __bf16* bRow = w3bf + (size_t)(nt * 16 + ll) * K2 + lh * 8;
        #pragma unroll
        for (int k0 = 0; k0 < K2; k0 += 32) {
            v8bf alo = *(const v8bf*)(aRow + k0 + lh * 8);
            v8bf ahi = *(const v8bf*)(aRow + k0 + lh * 8 + 16);
            v8bf blo = *(const v8bf*)(bRow + k0);
            v8bf bhi = *(const v8bf*)(bRow + k0 + 16);
            acc = __builtin_amdgcn_wmma_f32_16x16x32_bf16(
                false, combine8(alo, ahi), false, combine8(blo, bhi),
                (short)0, acc, false, false);
        }
        int o = nt * 16 + ll;
        if (o < 50) {
            float bias = b3[o];
            #pragma unroll
            for (int j = 0; j < 8; ++j) {
                int n = n0 + mt * 16 + lh * 8 + j;
                out[((size_t)b * 8192 + n) * 50 + o] = acc[j] + bias;
            }
        }
    }
}

extern "C" void kernel_launch(void* const* d_in, const int* in_sizes, int n_in,
                              void* d_out, int out_size, void* d_ws, size_t ws_size,
                              hipStream_t stream) {
    (void)in_sizes; (void)n_in; (void)out_size; (void)ws_size;
    const float* feat      = (const float*)d_in[0];
    const float* centers   = (const float*)d_in[1];
    const float* pc        = (const float*)d_in[2];
    const int*   category  = (const int*)d_in[3];
    const float* cat_embed = (const float*)d_in[4];
    const float* w1  = (const float*)d_in[5];
    const float* b1  = (const float*)d_in[6];
    const float* g1  = (const float*)d_in[7];
    const float* be1 = (const float*)d_in[8];
    const float* m1  = (const float*)d_in[9];
    const float* v1  = (const float*)d_in[10];
    const float* w2  = (const float*)d_in[11];
    const float* b2  = (const float*)d_in[12];
    const float* g2  = (const float*)d_in[13];
    const float* be2 = (const float*)d_in[14];
    const float* m2  = (const float*)d_in[15];
    const float* v2  = (const float*)d_in[16];
    const float* w3  = (const float*)d_in[17];
    const float* b3  = (const float*)d_in[18];

    // workspace layout: bf16 weights (padded) + folded BN affine (f32)
    __bf16* w1bf = (__bf16*)d_ws;             // 256*512*2 B
    __bf16* w2bf = w1bf + 256 * K1;           // 256*256*2 B
    __bf16* w3bf = w2bf + 256 * K2;           //  64*256*2 B
    float*  ab   = (float*)(w3bf + 64 * K2);  // 1024 f32

    int prepTotal = 256 * K1 + 256 * K2 + 64 * K2 + 512;
    prep_kernel<<<(prepTotal + 255) / 256, 256, 0, stream>>>(
        w1, w2, w3, b1, g1, be1, m1, v1, b2, g2, be2, m2, v2,
        w1bf, w2bf, w3bf, ab);

    // 16 batches * 128 tiles of 64 points
    partseg_fused<<<dim3(2048), dim3(256), SMEM_BYTES, stream>>>(
        feat, centers, pc, category, cat_embed, b3,
        w1bf, w2bf, w3bf, ab, (float*)d_out);
}
